// VectorQuantizer_55989193671483
// MI455X (gfx1250) — matrix operand order repaired and checked by reading, not compiled, not run on hardware
//
#include <hip/hip_runtime.h>
#include <hip/hip_bf16.h>

#define NUM_EMB 4096
#define EMB_DIM 256
#define N_VEC   65536                      // 16*64*64 rows
#define ROWS_PER_BLOCK 32
#define THREADS_MAIN 64                    // 2 waves (wave32)
#define K_TILE 16
#define N_TILES (NUM_EMB / K_TILE)         // 256

typedef __attribute__((ext_vector_type(16))) __bf16 v16bf;
typedef __attribute__((ext_vector_type(8)))  float  v8f;
typedef __attribute__((ext_vector_type(4)))  int    v4i;

#define AS1 __attribute__((address_space(1)))
#define AS3 __attribute__((address_space(3)))

#if defined(__has_builtin)
#  if __has_builtin(__builtin_amdgcn_global_load_async_to_lds_b128)
#    define HAVE_ASYNC 1
#  endif
#endif
#ifndef HAVE_ASYNC
#  define HAVE_ASYNC 0
#endif

// d_out layout (floats): quantized | loss | encodings | indices
#define Q_SIZE   (N_VEC * EMB_DIM)                 // 16777216
#define LOSS_OFF ((size_t)Q_SIZE)                  // 16777216
#define ENC_OFF  ((size_t)Q_SIZE + 1)              // 16777217
#define ENC_SIZE ((size_t)N_VEC * NUM_EMB)         // 268435456
#define IDX_OFF  (ENC_OFF + ENC_SIZE)              // 285212673

__device__ __forceinline__ void wait_async0() {
#if defined(__has_builtin) && __has_builtin(__builtin_amdgcn_s_wait_asynccnt)
    __builtin_amdgcn_s_wait_asynccnt(0);
#else
    asm volatile("s_wait_asynccnt 0x0" ::: "memory");
#endif
}

// ---------------- prep: f32 embedding -> bf16 hi/lo split tables ----------------
__global__ void vq_prep(const float* __restrict__ emb,
                        unsigned short* __restrict__ eh,
                        unsigned short* __restrict__ el,
                        float* __restrict__ loss_cell) {
    size_t i  = (size_t)blockIdx.x * blockDim.x + threadIdx.x;
    size_t st = (size_t)gridDim.x * blockDim.x;
    if (i == 0) *loss_cell = 0.0f;
    for (; i < (size_t)NUM_EMB * EMB_DIM; i += st) {
        float v = emb[i];
        unsigned u = __builtin_bit_cast(unsigned, v);
        unsigned short h = (unsigned short)(u >> 16);
        float hf = __builtin_bit_cast(float, u & 0xFFFF0000u);
        float lo = v - hf;
        unsigned short l = (unsigned short)(__builtin_bit_cast(unsigned, lo) >> 16);
        eh[i] = h;
        el[i] = l;
    }
}

// ---------------- per-code squared norm ----------------
__global__ void vq_enorm(const float* __restrict__ emb, float* __restrict__ enorm) {
    int code = blockIdx.x;
    int lane = threadIdx.x;           // blockDim 32 == one wave
    const float* r = emb + (size_t)code * EMB_DIM;
    float s = 0.0f;
    for (int j = lane; j < EMB_DIM; j += 32) { float v = r[j]; s += v * v; }
    #pragma unroll
    for (int o = 16; o; o >>= 1) s += __shfl_xor(s, o, 32);
    if (lane == 0) enorm[code] = s;
}

// ---------------- zero the one-hot region (every call) ----------------
__global__ void vq_zero(float* __restrict__ p, size_t n) {
    size_t i  = (size_t)blockIdx.x * blockDim.x + threadIdx.x;
    size_t st = (size_t)gridDim.x * blockDim.x;
    for (; i < n; i += st) p[i] = 0.0f;
}

// Stage one 16-code tile (hi+lo, fragment-major) into LDS buffer `bhi/blo`.
// Fragment (c,L): column n=L&15, K = (L>>4)*16 + e within chunk c -> 32B contiguous.
__device__ __forceinline__ void stage_tile(const __bf16* __restrict__ ehi,
                                           const __bf16* __restrict__ elo,
                                           v16bf* bhi, v16bf* blo,
                                           int cb, int tid) {
    for (int s = tid; s < 8 * 32; s += THREADS_MAIN) {
        int c = s >> 5, L = s & 31, nn = L & 15, hh = L >> 4;
        size_t off = (size_t)(cb + nn) * EMB_DIM + c * 32 + hh * 16;
#if HAVE_ASYNC
        __builtin_amdgcn_global_load_async_to_lds_b128(
            (AS1 v4i*)(ehi + off),     (AS3 v4i*)(bhi + s),                     0, 0);
        __builtin_amdgcn_global_load_async_to_lds_b128(
            (AS1 v4i*)(ehi + off + 8), (AS3 v4i*)((char*)(bhi + s) + 16),      0, 0);
        __builtin_amdgcn_global_load_async_to_lds_b128(
            (AS1 v4i*)(elo + off),     (AS3 v4i*)(blo + s),                     0, 0);
        __builtin_amdgcn_global_load_async_to_lds_b128(
            (AS1 v4i*)(elo + off + 8), (AS3 v4i*)((char*)(blo + s) + 16),      0, 0);
#else
        bhi[s] = *(const v16bf*)(ehi + off);
        blo[s] = *(const v16bf*)(elo + off);
#endif
    }
}

// ---------------- main: WMMA distance GEMM + argmin + outputs ----------------
__global__ __launch_bounds__(THREADS_MAIN) void vq_main(
    const float*  __restrict__ xin,
    const __bf16* __restrict__ ehi,
    const __bf16* __restrict__ elo,
    const float*  __restrict__ enorm,
    const float*  __restrict__ emb,
    float*        __restrict__ out)
{
    __shared__ v16bf BhiS[2][8 * 32];                   // 16 KB double-buffered B (hi)
    __shared__ v16bf BloS[2][8 * 32];                   // 16 KB double-buffered B (lo)
    __shared__ int   fidx[ROWS_PER_BLOCK];
    __shared__ float lsum_sh;

    const int tid  = threadIdx.x;
    const int lane = tid & 31;
    const int wave = tid >> 5;
    const int R0   = blockIdx.x * ROWS_PER_BLOCK;

    if (tid == 0) lsum_sh = 0.0f;

    // Build per-wave A fragments once (wave owns 16 rows). A 16x32 layout (matches B staging):
    // lanes 0-15 hold K {0..15}, lanes 16-31 hold K {16..31}, pairs per VGPR.
    const int half = lane >> 4;
    const int grow = R0 + wave * 16 + (lane & 15);      // this lane's global row
    const float* xrow = xin + (size_t)grow * EMB_DIM;
    v16bf Ahi[8], Alo[8];
    #pragma unroll
    for (int c = 0; c < 8; ++c) {
        union { v16bf v; __bf16 h[16]; } uh, ul;
        #pragma unroll
        for (int e = 0; e < 16; ++e) {
            int p = e >> 1, hw = e & 1;
            int k = half * 16 + 2 * p + hw;
            float xv = xrow[32 * c + k];
            unsigned u = __builtin_bit_cast(unsigned, xv);
            uh.h[e] = __builtin_bit_cast(__bf16, (unsigned short)(u >> 16));
            float hf = __builtin_bit_cast(float, u & 0xFFFF0000u);
            float lo = xv - hf;
            ul.h[e] = __builtin_bit_cast(__bf16,
                        (unsigned short)(__builtin_bit_cast(unsigned, lo) >> 16));
        }
        Ahi[c] = uh.v; Alo[c] = ul.v;
    }

    float vmin[8]; int vidx[8];
    #pragma unroll
    for (int i = 0; i < 8; ++i) { vmin[i] = 3.4e38f; vidx[i] = 0; }

    const int ncol = lane & 15;

    // prologue: stage tile 0 into buffer 0
    stage_tile(ehi, elo, BhiS[0], BloS[0], 0, tid);
    wait_async0();
    __syncthreads();

    for (int t = 0; t < N_TILES; ++t) {
        const int cur = t & 1;
        // issue async staging of next tile into the other buffer (overlaps WMMA below)
        if (t + 1 < N_TILES)
            stage_tile(ehi, elo, BhiS[1 - cur], BloS[1 - cur], (t + 1) * K_TILE, tid);

        // two independent accumulator chains: hi*hi and the two cross terms
        v8f acc0 = {};
        v8f acc1 = {};
        #pragma unroll
        for (int c = 0; c < 8; ++c) {
            v16bf bh = BhiS[cur][c * 32 + lane];
            v16bf bl = BloS[cur][c * 32 + lane];
            acc0 = __builtin_amdgcn_wmma_f32_16x16x32_bf16(false, Ahi[c], false, bh, (short)0, acc0, false, false);
            acc1 = __builtin_amdgcn_wmma_f32_16x16x32_bf16(false, Ahi[c], false, bl, (short)0, acc1, false, false);
            acc1 = __builtin_amdgcn_wmma_f32_16x16x32_bf16(false, Alo[c], false, bh, (short)0, acc1, false, false);
        }

        const int  cb   = t * K_TILE;
        const int  code = cb + ncol;
        const float en  = enorm[code];
        #pragma unroll
        for (int i = 0; i < 8; ++i) {
            float s = en - 2.0f * (acc0[i] + acc1[i]);  // ||x||^2 constant per row: dropped
            if (s < vmin[i]) { vmin[i] = s; vidx[i] = code; }
        }

        if (t + 1 < N_TILES) wait_async0();             // next tile landed in LDS
        __syncthreads();
    }

    // reduce across the 16 columns (xor masks 1..8 stay within each 16-lane half)
    #pragma unroll
    for (int off = 1; off < 16; off <<= 1) {
        #pragma unroll
        for (int i = 0; i < 8; ++i) {
            float om = __shfl_xor(vmin[i], off, 32);
            int   oi = __shfl_xor(vidx[i], off, 32);
            if (om < vmin[i] || (om == vmin[i] && oi < vidx[i])) { vmin[i] = om; vidx[i] = oi; }
        }
    }
    // C layout: slot i -> row i (lanes 0-15) / row 8+i (lanes 16-31)
    if ((lane & 15) == 0) {
        int base = wave * 16 + half * 8;
        #pragma unroll
        for (int i = 0; i < 8; ++i) fidx[base + i] = vidx[i];
    }
    __syncthreads();

    // epilogue: indices + one-hot scatter
    float* out_enc = out + ENC_OFF;
    float* out_idx = out + IDX_OFF;
    for (int r = tid; r < ROWS_PER_BLOCK; r += THREADS_MAIN) {
        int idx = fidx[r];
        out_idx[R0 + r] = (float)idx;
        out_enc[(size_t)(R0 + r) * NUM_EMB + idx] = 1.0f;
    }

    // quantized gather (== straight-through output) + loss partial (x re-read from HBM)
    float lsum = 0.0f;
    for (int i = tid; i < ROWS_PER_BLOCK * (EMB_DIM / 4); i += THREADS_MAIN) {
        int r = i >> 6, c4 = i & 63;
        int idx = fidx[r];
        float4 q = *(const float4*)(emb + (size_t)idx * EMB_DIM + c4 * 4);
        *(float4*)(out + (size_t)(R0 + r) * EMB_DIM + c4 * 4) = q;
        float4 xv = *(const float4*)(xin + (size_t)(R0 + r) * EMB_DIM + c4 * 4);
        float dx = q.x - xv.x, dy = q.y - xv.y, dz = q.z - xv.z, dw = q.w - xv.w;
        lsum += dx * dx + dy * dy + dz * dz + dw * dw;
    }
    #pragma unroll
    for (int off = 16; off; off >>= 1) lsum += __shfl_xor(lsum, off, 32);
    if (lane == 0) atomicAdd(&lsum_sh, lsum);
    __syncthreads();
    if (tid == 0)
        atomicAdd(out + LOSS_OFF, lsum_sh * (1.25f / ((float)N_VEC * (float)EMB_DIM)));
}

extern "C" void kernel_launch(void* const* d_in, const int* in_sizes, int n_in,
                              void* d_out, int out_size, void* d_ws, size_t ws_size,
                              hipStream_t stream) {
    (void)in_sizes; (void)n_in; (void)out_size; (void)ws_size;
    const float* xin = (const float*)d_in[0];
    const float* emb = (const float*)d_in[1];
    float* out = (float*)d_out;

    unsigned short* eh = (unsigned short*)d_ws;                           // 2 MB
    unsigned short* el = (unsigned short*)((char*)d_ws + (2u << 20));     // 2 MB
    float*       enorm = (float*)((char*)d_ws + (4u << 20));              // 16 KB

    vq_prep <<<1024, 256, 0, stream>>>(emb, eh, el, out + LOSS_OFF);
    vq_enorm<<<NUM_EMB, 32, 0, stream>>>(emb, enorm);
    vq_zero <<<8192, 256, 0, stream>>>(out + ENC_OFF, ENC_SIZE);
    vq_main <<<N_VEC / ROWS_PER_BLOCK, THREADS_MAIN, 0, stream>>>(
        xin, (const __bf16*)eh, (const __bf16*)el, enorm, emb, out);
}
